// SinkhornDistance_69990787056127
// MI455X (gfx1250) — compile-verified, matrix operand order checked by reading, not run on hardware
//
#include <hip/hip_runtime.h>
#include <hip/hip_bf16.h>
#include <math.h>

// Sinkhorn distance, MI455X (gfx1250, wave32).
// C = |x_i - y_j|^2 built via split-f16 WMMA (f32-accurate), Sinkhorn sweeps
// are L2-resident streaming logsumexp passes, everything deterministic.

#define NPT 4096
#define DIM 64
#define EPSV 0.1f
#define INV_EPS 10.0f
#define MAX_ITER 100
#define THRESH 0.1f
#define LOG_MU (-8.3177252f)   // log(1/4096 + 1e-8)

// workspace layout, in 4-byte units
#define WXN    0        // x row norms   [4096]
#define WYN    4096     // y row norms   [4096]
#define WU     8192     // u (committed) [4096]
#define WUN    12288    // u_new staging [4096]
#define WV     16384    // v (committed) [4096]
#define WERRB  20480    // per-row |u_new-u| [4096]
#define WCOSTB 24576    // per-row cost partials [4096]
#define WDONE  28672    // int convergence flag

typedef __attribute__((ext_vector_type(16))) _Float16 v16h;
typedef __attribute__((ext_vector_type(8)))  float    v8f;

// ---------------------------------------------------------------- init ----
__global__ __launch_bounds__(256) void k_init(const float* __restrict__ x,
                                              const float* __restrict__ y,
                                              float* __restrict__ ws,
                                              float* __restrict__ out) {
  int t = blockIdx.x * blockDim.x + threadIdx.x;
  if (t < NPT) {
    float sx = 0.f, sy = 0.f;
    #pragma unroll
    for (int i = 0; i < DIM; ++i) {
      float a = x[t * DIM + i]; sx += a * a;
      float b = y[t * DIM + i]; sy += b * b;
    }
    ws[WXN + t] = sx;  ws[WYN + t] = sy;
    ws[WU + t] = 0.f;  ws[WUN + t] = 0.f;  ws[WV + t] = 0.f;
    ws[WERRB + t] = 0.f;  ws[WCOSTB + t] = 0.f;
  }
  if (t == 0) { ((int*)ws)[WDONE] = 0; out[0] = 0.f; }
}

// ------------------------------------------------- cost matrix via WMMA ----
// C = xnorm_i + ynorm_j - 2 * (x @ y^T). One wave per 16x16 tile.
// f32 inputs split into hi/lo f16 halves; S = Ah*Bh + Ah*Bl + Al*Bh accumulated
// in f32 by v_wmma_f32_16x16x32_f16 -> ~f32 accuracy at the fast-f16 rate.
__global__ __launch_bounds__(256) void k_cost(const float* __restrict__ x,
                                              const float* __restrict__ y,
                                              const float* __restrict__ ws,
                                              float* __restrict__ Cm) {
  const int lane = threadIdx.x & 31;
  const int wave = threadIdx.x >> 5;
  const int tile = blockIdx.x * 8 + wave;
  const int tm = (tile >> 8) << 4;       // 256 tiles per dim
  const int tn = (tile & 255) << 4;
  const int r = lane & 15;
  const int h = lane >> 4;               // half-wave selects K sub-chunk

  const float* xrow = x + (size_t)(tm + r) * DIM;
  const float* yrow = y + (size_t)(tn + r) * DIM;

  v8f acc = {};
  #pragma unroll
  for (int k0 = 0; k0 < DIM; k0 += 32) {
    const int kb = k0 + h * 8;           // 16-bit A/B layout: K=kb..kb+7, kb+16..kb+23
    float xf[16], yf[16];
    #pragma unroll
    for (int i = 0; i < 8; ++i) {
      xf[i]     = xrow[kb + i];      xf[8 + i] = xrow[kb + 16 + i];
      yf[i]     = yrow[kb + i];      yf[8 + i] = yrow[kb + 16 + i];
    }
    v16h ah, al, bh, bl;
    #pragma unroll
    for (int i = 0; i < 16; ++i) {
      _Float16 xh = (_Float16)xf[i];
      ah[i] = xh;  al[i] = (_Float16)(xf[i] - (float)xh);
      _Float16 yh = (_Float16)yf[i];
      bh[i] = yh;  bl[i] = (_Float16)(yf[i] - (float)yh);
    }
    acc = __builtin_amdgcn_wmma_f32_16x16x32_f16(false, al, false, bh, (short)0, acc, false, false);
    acc = __builtin_amdgcn_wmma_f32_16x16x32_f16(false, ah, false, bl, (short)0, acc, false, false);
    acc = __builtin_amdgcn_wmma_f32_16x16x32_f16(false, ah, false, bh, (short)0, acc, false, false);
  }

  const float yn = ws[WYN + tn + r];
  #pragma unroll
  for (int vg = 0; vg < 8; ++vg) {
    const int row = tm + vg + h * 8;     // D layout: vgpr=row, lane16-31 -> +8
    const int col = tn + r;
    float c = ws[WXN + row] + yn - 2.0f * acc[vg];
    Cm[(size_t)row * NPT + col] = c;
  }
}

// ---------------------------------------------------------- helpers -------
__device__ __forceinline__ void lse_push(float a, float& m, float& s) {
  float mn = fmaxf(m, a);
  s = s * __expf(m - mn) + __expf(a - mn);
  m = mn;
}

// -------------------------------------------------- row pass: u_new -------
// u_new_i = eps*(log_mu - LSE_j((v_j - C_ij)/eps)); one wave per row.
__global__ __launch_bounds__(256) void k_row(const float* __restrict__ Cm,
                                             float* __restrict__ ws) {
  if (((const int*)ws)[WDONE]) return;
  const int lane = threadIdx.x & 31;
  const int row = blockIdx.x * 8 + (threadIdx.x >> 5);
  const float* crow = Cm + (size_t)row * NPT;

  float m = -INFINITY, s = 0.f;
  #pragma unroll 4
  for (int j = lane; j < NPT; j += 32)
    lse_push((ws[WV + j] - crow[j]) * INV_EPS, m, s);

  #pragma unroll
  for (int off = 16; off > 0; off >>= 1) {
    float mo = __shfl_xor(m, off, 32);
    float so = __shfl_xor(s, off, 32);
    float mn = fmaxf(m, mo);
    s = s * __expf(m - mn) + so * __expf(mo - mn);
    m = mn;
  }
  if (lane == 0) {
    float un = EPSV * (LOG_MU - (m + __logf(s)));
    ws[WUN + row] = un;
    ws[WERRB + row] = fabsf(un - ws[WU + row]);
  }
}

// ------------------------------------------- col pass: v_new + commit u ----
// v_new_j = eps*(log_nu - LSE_i((u_new_i - C_ij)/eps)); block = 32 cols,
// 8 waves split the i range (coalesced row-major reads), merge in LDS.
__global__ __launch_bounds__(256) void k_col(const float* __restrict__ Cm,
                                             float* __restrict__ ws) {
  if (((const int*)ws)[WDONE]) return;
  __shared__ float lm[8][32], ls[8][32];
  const int lane = threadIdx.x & 31;
  const int w = threadIdx.x >> 5;
  const int j = blockIdx.x * 32 + lane;
  const int i0 = w * (NPT / 8);

  float m = -INFINITY, s = 0.f;
  #pragma unroll 4
  for (int i = i0; i < i0 + NPT / 8; ++i)
    lse_push((ws[WUN + i] - Cm[(size_t)i * NPT + j]) * INV_EPS, m, s);

  lm[w][lane] = m;  ls[w][lane] = s;
  __syncthreads();
  if (threadIdx.x < 32) {
    float M = lm[0][lane], S = ls[0][lane];
    #pragma unroll
    for (int ww = 1; ww < 8; ++ww) {
      float mo = lm[ww][lane], so = ls[ww][lane];
      float mn = fmaxf(M, mo);
      S = S * __expf(M - mn) + so * __expf(mo - mn);
      M = mn;
    }
    ws[WV + j] = EPSV * (LOG_MU - (M + __logf(S)));   // log_nu == log_mu
    ws[WU + j] = ws[WUN + j];                         // commit u (any partition)
  }
}

// --------------------------------------- convergence flag (deterministic) --
__global__ __launch_bounds__(256) void k_flag(float* __restrict__ ws) {
  if (((const int*)ws)[WDONE]) return;
  __shared__ float red[256];
  float s = 0.f;
  for (int t = threadIdx.x; t < NPT; t += 256) s += ws[WERRB + t];
  red[threadIdx.x] = s;
  __syncthreads();
  for (int off = 128; off > 0; off >>= 1) {
    if (threadIdx.x < off) red[threadIdx.x] += red[threadIdx.x + off];
    __syncthreads();
  }
  if (threadIdx.x == 0 && red[0] < THRESH) ((int*)ws)[WDONE] = 1;
}

// --------------------------------- pi = exp(M), per-row cost partials -----
__global__ __launch_bounds__(256) void k_pi(const float* __restrict__ Cm,
                                            const float* __restrict__ ws,
                                            float* __restrict__ pi,
                                            float* __restrict__ costb) {
  __shared__ float red[256];
  const int row = blockIdx.x;
  const float ui = ws[WU + row];
  const float* crow = Cm + (size_t)row * NPT;
  float partial = 0.f;
  #pragma unroll
  for (int t = 0; t < NPT / 256; ++t) {
    int j = threadIdx.x + t * 256;
    float c = crow[j];
    float p = __expf((ui + ws[WV + j] - c) * INV_EPS);
    pi[(size_t)row * NPT + j] = p;
    partial += p * c;
  }
  red[threadIdx.x] = partial;
  __syncthreads();
  for (int off = 128; off > 0; off >>= 1) {
    if (threadIdx.x < off) red[threadIdx.x] += red[threadIdx.x + off];
    __syncthreads();
  }
  if (threadIdx.x == 0) costb[row] = red[0];
}

__global__ __launch_bounds__(256) void k_cost_reduce(const float* __restrict__ costb,
                                                     float* __restrict__ out) {
  __shared__ float red[256];
  float s = 0.f;
  for (int t = threadIdx.x; t < NPT; t += 256) s += costb[t];
  red[threadIdx.x] = s;
  __syncthreads();
  for (int off = 128; off > 0; off >>= 1) {
    if (threadIdx.x < off) red[threadIdx.x] += red[threadIdx.x + off];
    __syncthreads();
  }
  if (threadIdx.x == 0) out[0] = red[0];
}

// ------------------------------------------------------------- launch -----
extern "C" void kernel_launch(void* const* d_in, const int* in_sizes, int n_in,
                              void* d_out, int out_size, void* d_ws, size_t ws_size,
                              hipStream_t stream) {
  (void)in_sizes; (void)n_in; (void)out_size; (void)ws_size;
  const float* x = (const float*)d_in[0];
  const float* y = (const float*)d_in[1];
  float* out = (float*)d_out;
  float* ws  = (float*)d_ws;
  float* pi  = out + 1;                          // output order: cost, pi, C
  float* Cm  = out + 1 + (size_t)NPT * NPT;

  k_init<<<NPT / 256, 256, 0, stream>>>(x, y, ws, out);
  k_cost<<<(256 * 256) / 8, 256, 0, stream>>>(x, y, ws, Cm);   // 65536 tiles / 8 waves
  for (int it = 0; it < MAX_ITER; ++it) {
    k_row<<<NPT / 8, 256, 0, stream>>>(Cm, ws);
    k_col<<<NPT / 32, 256, 0, stream>>>(Cm, ws);
    k_flag<<<1, 256, 0, stream>>>(ws);
  }
  k_pi<<<NPT, 256, 0, stream>>>(Cm, ws, pi, ws + WCOSTB);
  k_cost_reduce<<<1, 256, 0, stream>>>(ws + WCOSTB, out);
}
